// ExpireSpanTransformerXL_39865886441793
// MI455X (gfx1250) — compile-verified, matrix operand order checked by reading, not run on hardware
//
#include <hip/hip_runtime.h>
#include <hip/hip_bf16.h>

// ---------------------------------------------------------------------------
// Model constants (from the reference): L=4, B=2, N=1024, D=1024, H=16,
// V=32000, M=2048, DH=64, FF=4096
// ---------------------------------------------------------------------------
#define LL   4
#define BB   2
#define NNc  1024
#define DD   1024
#define HH   16
#define VV   32000
#define MMc  2048
#define DHH  64
#define FFD  4096
#define CTX  (MMc + NNc)
#define SCALEF 0.125f        // DH^-0.5
#define MAXMEMF 2048.0f
#define RAMPF   128.0f
#define AUXW    (1e-06f / 1024.0f)   // COEF / SEQ

typedef __attribute__((ext_vector_type(16))) __bf16 v16bf;
typedef __attribute__((ext_vector_type(8)))  __bf16 v8bfv;
typedef __attribute__((ext_vector_type(8)))  float  v8f;

// ---------------------------------------------------------------------------
// CDNA5 async global->LDS helpers (ASYNCcnt path, portable via inline asm)
// ---------------------------------------------------------------------------
typedef __attribute__((address_space(3))) char lds_char_t;

__device__ __forceinline__ unsigned ldsoff(const void* p)
{
    // generic -> LDS addrspace cast, then ptrtoint: raw LDS byte offset
    return (unsigned)(unsigned long long)(lds_char_t*)p;
}

__device__ __forceinline__ void async_ld_b128(unsigned lds, const void* gaddr)
{
    asm volatile("global_load_async_to_lds_b128 %0, %1, off"
                 :: "v"(lds), "v"((unsigned long long)gaddr)
                 : "memory");
}

__device__ __forceinline__ void wait_async0()
{
    asm volatile("s_wait_asynccnt 0" ::: "memory");
}

__device__ __forceinline__ void wait_ds0()
{
    asm volatile("s_wait_dscnt 0" ::: "memory");
}

// ---------------------------------------------------------------------------
// Generic tiled bf16 WMMA GEMM:  C[M x N] = act(A[M x K] @ W[K x N] + bias) + resid
// Block = 256 threads (8 waves), C tile 128x64, each wave 32x32 (2x2 WMMA tiles).
// K-step 64: A tile async-copied straight into LDS (ASYNCcnt), B tile staged
// transposed; 8 WMMAs per barrier pair. Optional batched mode for (b,h) GEMMs.
// ---------------------------------------------------------------------------
__global__ __launch_bounds__(256)
void k_gemm(const __bf16* __restrict__ A, long lda,
            const __bf16* __restrict__ W, long ldw,
            const float* __restrict__ bias,
            const float* __restrict__ resid, long ldr,
            float* Cf, __bf16* Cb, long ldc,
            int K, int act_gelu,
            int batchH, long sAb, long sAh, long sC)
{
    const long z = blockIdx.z;
    if (batchH > 0) {
        A += (z / batchH) * sAb + (z % batchH) * sAh;
        if (Cf)    Cf    += z * sC;
        if (Cb)    Cb    += z * sC;
        if (resid) resid += z * sC;
    }
    __shared__ alignas(128) __bf16 As[128][64];
    __shared__ alignas(128) __bf16 Bs[64][64];   // stored transposed: Bs[n][k]

    const int tid  = threadIdx.x;
    const int lane = tid & 31;
    const int wave = tid >> 5;
    const int wrow = wave >> 1;      // 0..3
    const int wcol = wave & 1;       // 0..1
    const int lh   = lane >> 4;      // 0/1 half
    const int ll   = lane & 15;
    const long row0 = (long)blockIdx.y * 128;
    const long col0 = (long)blockIdx.x * 64;

    // staging assignments
    const int ar = tid >> 1;             // A row 0..127 (64B per thread)
    const int ac = (tid & 1) * 32;       // A col 0/32
    const unsigned aoff = ldsoff(&As[ar][ac]);
    const int bk = tid >> 2;             // B k-row 0..63 (16 elems per thread)
    const int bn = (tid & 3) * 16;       // B n col 0/16/32/48

    v8f acc[2][2] = {};

    for (int k0 = 0; k0 < K; k0 += 64) {
        // ---- A tile (128 x 64): async DMA global -> LDS, 4 x 16B per thread
        const __bf16* gp = A + (row0 + ar) * lda + k0 + ac;
#pragma unroll
        for (int u = 0; u < 4; ++u)
            async_ld_b128(aoff + u * 16, gp + u * 8);

        // ---- B tile (64 x 64) transposed into Bs[n][k]
        {
            v16bf wv = *(const v16bf*)(W + (long)(k0 + bk) * ldw + col0 + bn);
#pragma unroll
            for (int j = 0; j < 16; ++j) Bs[bn + j][bk] = wv[j];
            if (k0 + 64 < K)
                __builtin_prefetch((const void*)(W + (long)(k0 + 64 + bk) * ldw + col0 + bn), 0, 1);
        }
        wait_async0();
        __syncthreads();

#pragma unroll
        for (int kc = 0; kc < 2; ++kc) {
            v16bf af[2], bfr[2];
#pragma unroll
            for (int mi = 0; mi < 2; ++mi)
                af[mi] = *(const v16bf*)&As[wrow * 32 + mi * 16 + ll][kc * 32 + lh * 16];
#pragma unroll
            for (int ni = 0; ni < 2; ++ni)
                bfr[ni] = *(const v16bf*)&Bs[wcol * 32 + ni * 16 + ll][kc * 32 + lh * 16];
#pragma unroll
            for (int mi = 0; mi < 2; ++mi)
#pragma unroll
                for (int ni = 0; ni < 2; ++ni)
                    acc[mi][ni] = __builtin_amdgcn_wmma_f32_16x16x32_bf16(
                        false, af[mi], false, bfr[ni], (short)0, acc[mi][ni], false, false);
        }
        __syncthreads();
    }

    // ---- epilogue: bias -> gelu -> residual -> store f32/bf16
#pragma unroll
    for (int mi = 0; mi < 2; ++mi)
#pragma unroll
        for (int ni = 0; ni < 2; ++ni)
#pragma unroll
            for (int g = 0; g < 8; ++g) {
                const long r = row0 + wrow * 32 + mi * 16 + g + 8 * lh;
                const long c = col0 + wcol * 32 + ni * 16 + ll;
                float v = acc[mi][ni][g];
                if (bias)     v += bias[c];
                if (act_gelu) v = 0.5f * v * (1.0f + erff(v * 0.70710678f));
                if (resid)    v += resid[r * ldr + c];
                if (Cf) Cf[r * ldc + c] = v;
                if (Cb) Cb[r * ldc + c] = (__bf16)v;
            }
}

// ---------------------------------------------------------------------------
// Flash attention with WMMA. Block = 128 threads (4 waves); each wave owns a
// 16-row Q stripe for one (b,h); streams 32 keys per iteration.
// V tile async-DMAd to LDS at the top of each iteration (overlaps the S WMMAs
// and softmax VALU); S = Q Kt (wmma), rel-shift pos bias, causal mask, online
// softmax, O += (P * emask) V (wmma, P bounced through LDS for layout).
// Waves are fully independent (per-wave ASYNCcnt/DScnt waits, no barriers).
// ---------------------------------------------------------------------------
__global__ __launch_bounds__(128)
void k_attn(const __bf16* __restrict__ qb,     // (B*N, D)
            const __bf16* __restrict__ kvb,    // (B*CTX, 2D): k cols [0,D), v cols [D,2D)
            const float*  __restrict__ qpos,   // (B*H, N, N)
            const float*  __restrict__ emask,  // (B, CTX)
            __bf16*       __restrict__ outb)   // (B*N, D)
{
    const int bh = blockIdx.y;
    const int b  = bh >> 4;          // H = 16
    const int hh = bh & 15;
    const int lane = threadIdx.x & 31;
    const int w    = threadIdx.x >> 5;
    const int i0   = blockIdx.x * 64 + w * 16;
    const int lh   = lane >> 4, ll = lane & 15;

    __shared__ alignas(128) __bf16 P[4][16][32];    // per-wave P tile
    __shared__ alignas(128) __bf16 Vs[4][32][64];   // per-wave V tile

    // Q fragments (16 x 64 -> two K=32 A-fragments)
    v16bf aq[2];
#pragma unroll
    for (int f = 0; f < 2; ++f)
        aq[f] = *(const v16bf*)(qb + ((long)(b * NNc + i0 + ll)) * DD + hh * 64 + f * 32 + lh * 16);

    v8f o[4] = {};
    float m[8], ls[8];
#pragma unroll
    for (int g = 0; g < 8; ++g) { m[g] = -INFINITY; ls[g] = 0.f; }

    const int jmax = i0 + 15 + MMc;
    for (int jj0 = 0; jj0 <= jmax; jj0 += 32) {
        // ---- async-stage V tile (32 keys x 64 dh) for this iteration
        {
            const __bf16* vb = kvb + ((long)(b * CTX + jj0)) * (2 * DD) + DD + hh * 64;
#pragma unroll
            for (int q8 = 0; q8 < 8; ++q8) {
                const int chunk = q8 * 32 + lane;      // 0..255 16B chunks
                const int row = chunk >> 3;            // key row 0..31
                const int cc  = (chunk & 7) * 8;       // dh col
                async_ld_b128(ldsoff(&Vs[w][row][cc]), vb + (long)row * (2 * DD) + cc);
            }
        }

        // ---- scores S (16 x 32) = Q @ K^T, two 16x16 C fragments
        v8f s0 = {}, s1 = {};
#pragma unroll
        for (int f = 0; f < 2; ++f) {
            v16bf bk0 = *(const v16bf*)(kvb + ((long)(b * CTX + jj0 + ll)) * (2 * DD)
                                        + hh * 64 + f * 32 + lh * 16);
            v16bf bk1 = *(const v16bf*)(kvb + ((long)(b * CTX + jj0 + 16 + ll)) * (2 * DD)
                                        + hh * 64 + f * 32 + lh * 16);
            s0 = __builtin_amdgcn_wmma_f32_16x16x32_bf16(false, aq[f], false, bk0, (short)0, s0, false, false);
            s1 = __builtin_amdgcn_wmma_f32_16x16x32_bf16(false, aq[f], false, bk1, (short)0, s1, false, false);
        }

        const int j0c = jj0 + ll, j1c = jj0 + 16 + ll;
        const float em0 = emask[(long)b * CTX + j0c];
        const float em1 = emask[(long)b * CTX + j1c];

#pragma unroll
        for (int g = 0; g < 8; ++g) {
            const int i = i0 + g + 8 * lh;
            float v0 = s0[g] * SCALEF, v1 = s1[g] * SCALEF;
            // rel-shift position bias: valid for M <= jj <= i+M, row idx jj-M-i+N-1
            if (j0c >= MMc && j0c <= i + MMc)
                v0 += qpos[((long)bh * NNc + i) * NNc + (j0c - MMc - i + NNc - 1)] * SCALEF;
            if (j1c >= MMc && j1c <= i + MMc)
                v1 += qpos[((long)bh * NNc + i) * NNc + (j1c - MMc - i + NNc - 1)] * SCALEF;
            if (j0c > i + MMc) v0 = -INFINITY;
            if (j1c > i + MMc) v1 = -INFINITY;

            // row max across the 16 lanes of this half
            float r = fmaxf(v0, v1);
            r = fmaxf(r, __shfl_xor(r, 1, 32));
            r = fmaxf(r, __shfl_xor(r, 2, 32));
            r = fmaxf(r, __shfl_xor(r, 4, 32));
            r = fmaxf(r, __shfl_xor(r, 8, 32));
            const float mn = fmaxf(m[g], r);
            const float corr = (mn == -INFINITY) ? 1.f : __expf(m[g] - mn);
            m[g] = mn;
            ls[g] *= corr;
#pragma unroll
            for (int t = 0; t < 4; ++t) o[t][g] *= corr;
            const float p0 = (mn == -INFINITY) ? 0.f : __expf(v0 - mn);
            const float p1 = (mn == -INFINITY) ? 0.f : __expf(v1 - mn);
            ls[g] += p0 + p1;   // denominator WITHOUT emask (softmax then *emask)
            P[w][g + 8 * lh][ll]      = (__bf16)(p0 * em0);
            P[w][g + 8 * lh][16 + ll] = (__bf16)(p1 * em1);
        }
        wait_ds0();       // wave-local LDS turnaround for P
        wait_async0();    // V tile landed in LDS

        // ---- O += P @ V ; P as A-fragment (16x32), V as four 32x16 B-fragments
        v16bf ap = *(const v16bf*)&P[w][ll][lh * 16];
#pragma unroll
        for (int t = 0; t < 4; ++t) {
            v16bf bv;
#pragma unroll
            for (int tt = 0; tt < 16; ++tt)
                bv[tt] = Vs[w][lh * 16 + tt][t * 16 + ll];
            o[t] = __builtin_amdgcn_wmma_f32_16x16x32_bf16(false, ap, false, bv, (short)0, o[t], false, false);
        }
    }

    // ---- finalize: reduce denominator, normalize, store bf16
#pragma unroll
    for (int g = 0; g < 8; ++g) {
        float s = ls[g];
        s += __shfl_xor(s, 1, 32);
        s += __shfl_xor(s, 2, 32);
        s += __shfl_xor(s, 4, 32);
        s += __shfl_xor(s, 8, 32);
        const float inv = (s > 0.f) ? 1.f / s : 0.f;
        const int i = i0 + g + 8 * lh;
        const long base = ((long)(b * NNc + i)) * DD + hh * 64;
#pragma unroll
        for (int t = 0; t < 4; ++t)
            outb[base + t * 16 + ll] = (__bf16)(o[t][g] * inv);
    }
}

// ---------------------------------------------------------------------------
// Elementwise / small kernels
// ---------------------------------------------------------------------------
__global__ void k_zero1(float* p) { if (threadIdx.x == 0) p[0] = 0.f; }

__global__ __launch_bounds__(256)
void k_cast(const float* __restrict__ s, __bf16* __restrict__ d, long n)
{
    const long i = ((long)blockIdx.x * 256 + threadIdx.x) * 4;
    if (i >= n) return;
    float4 v = *(const float4*)(s + i);
    d[i + 0] = (__bf16)v.x; d[i + 1] = (__bf16)v.y;
    d[i + 2] = (__bf16)v.z; d[i + 3] = (__bf16)v.w;
}

__global__ __launch_bounds__(256)
void k_embed(const int* __restrict__ x, const float* __restrict__ emb, float* __restrict__ h)
{
    const long row = blockIdx.x;
    const long tok = x[row];
    const int c = threadIdx.x * 4;
    *(float4*)(h + row * DD + c) = *(const float4*)(emb + tok * DD + c);
}

__global__ __launch_bounds__(256)
void k_layernorm(const float* __restrict__ x, const float* __restrict__ gg,
                 const float* __restrict__ bb, __bf16* __restrict__ y)
{
    __shared__ float red[256];
    const long row = blockIdx.x;
    const int tid = threadIdx.x;
    float4 v = *(const float4*)(x + row * DD + tid * 4);
    red[tid] = v.x + v.y + v.z + v.w;
    __syncthreads();
    for (int s = 128; s > 0; s >>= 1) { if (tid < s) red[tid] += red[tid + s]; __syncthreads(); }
    const float mu = red[0] * (1.f / DD);
    __syncthreads();
    const float dx = v.x - mu, dy = v.y - mu, dz = v.z - mu, dw = v.w - mu;
    red[tid] = dx * dx + dy * dy + dz * dz + dw * dw;
    __syncthreads();
    for (int s = 128; s > 0; s >>= 1) { if (tid < s) red[tid] += red[tid + s]; __syncthreads(); }
    const float rs = rsqrtf(red[0] * (1.f / DD) + 1e-5f);
    const int c = tid * 4;
    y[row * DD + c + 0] = (__bf16)(dx * rs * gg[c + 0] + bb[c + 0]);
    y[row * DD + c + 1] = (__bf16)(dy * rs * gg[c + 1] + bb[c + 1]);
    y[row * DD + c + 2] = (__bf16)(dz * rs * gg[c + 2] + bb[c + 2]);
    y[row * DD + c + 3] = (__bf16)(dw * rs * gg[c + 3] + bb[c + 3]);
}

__global__ __launch_bounds__(256)
void k_ctx(const float* __restrict__ mem, const __bf16* __restrict__ xn, __bf16* __restrict__ ctx)
{
    const long row = blockIdx.x;
    const int b = (int)(row / CTX), jj = (int)(row % CTX);
    __bf16* dst = ctx + row * DD;
    if (jj < MMc) {
        const float* src = mem + ((long)b * MMc + jj) * DD;
        for (int i = threadIdx.x; i < DD; i += 256) dst[i] = (__bf16)src[i];
    } else {
        const __bf16* src = xn + ((long)b * NNc + (jj - MMc)) * DD;
        for (int i = threadIdx.x; i < DD; i += 256) dst[i] = src[i];
    }
}

// expire-span: exps = sigmoid(mem . Wexp + bexp)*MAXMEM ; emask; aux accumulation
__global__ __launch_bounds__(256)
void k_exps(const float* __restrict__ mem, const int* __restrict__ times,
            const float* __restrict__ Wexp, const float* __restrict__ bexp, int l,
            float* __restrict__ emask, float* __restrict__ aux)
{
    __shared__ float red[256];
    const long idx = blockIdx.x;                  // 0..B*CTX-1
    const int b = (int)(idx / CTX), jj = (int)(idx % CTX);
    const int tid = threadIdx.x;
    if (jj >= MMc) { if (tid == 0) emask[idx] = 1.f; return; }
    const float* mr = mem + ((long)b * MMc + jj) * DD;
    const int c = tid * 4;
    float4 mv = *(const float4*)(mr + c);
    float4 wv = *(const float4*)(Wexp + c);
    red[tid] = mv.x * wv.x + mv.y * wv.y + mv.z * wv.z + mv.w * wv.w;
    __syncthreads();
    for (int s = 128; s > 0; s >>= 1) { if (tid < s) red[tid] += red[tid + s]; __syncthreads(); }
    if (tid == 0) {
        const float z = 1.f / (1.f + __expf(-(red[0] + bexp[l]))) * MAXMEMF;
        const float t = (float)times[(long)b * MMc + jj];
        float em = (z - t) / RAMPF + 1.f;
        em = fminf(fmaxf(em, 0.f), 1.f);
        emask[idx] = em;
        if (em > 0.f && em < 1.f) atomicAdd(aux, z * AUXW);
    }
}

__global__ __launch_bounds__(256)
void k_sinusoid(__bf16* __restrict__ pe)
{
    const int row = blockIdx.x;
    const float t = (float)(NNc - 1 - row);
    for (int i = threadIdx.x; i < DD / 2; i += 256) {
        const float inv = __powf(10000.f, -2.f * (float)i / (float)DD);
        const float si = t * inv;
        pe[(long)row * DD + i]           = (__bf16)__sinf(si);
        pe[(long)row * DD + DD / 2 + i]  = (__bf16)__cosf(si);
    }
}

__global__ __launch_bounds__(256)
void k_posT(const __bf16* __restrict__ pos, __bf16* __restrict__ posT)
{
    const int d = blockIdx.y;
    const int t = blockIdx.x * 256 + threadIdx.x;
    posT[(long)d * NNc + t] = pos[(long)t * DHH + d];
}

// ---------------------------------------------------------------------------
// Host-side orchestration
// ---------------------------------------------------------------------------
extern "C" void kernel_launch(void* const* d_in, const int* in_sizes, int n_in,
                              void* d_out, int out_size, void* d_ws, size_t ws_size,
                              hipStream_t stream)
{
    const int*   x     = (const int*)  d_in[0];
    const float* mems  = (const float*)d_in[1];
    const int*   times = (const int*)  d_in[2];
    const float* emb   = (const float*)d_in[3];
    const float* Wq    = (const float*)d_in[4];
    const float* bq    = (const float*)d_in[5];
    const float* Wkv   = (const float*)d_in[6];
    const float* bkv   = (const float*)d_in[7];
    const float* Wo    = (const float*)d_in[8];
    const float* bo    = (const float*)d_in[9];
    const float* Wpos  = (const float*)d_in[10];
    const float* bpos  = (const float*)d_in[11];
    const float* Wexp  = (const float*)d_in[12];
    const float* bexp  = (const float*)d_in[13];
    const float* ln1g  = (const float*)d_in[14];
    const float* ln1b  = (const float*)d_in[15];
    const float* ln2g  = (const float*)d_in[16];
    const float* ln2b  = (const float*)d_in[17];
    const float* Wff1  = (const float*)d_in[18];
    const float* bff1  = (const float*)d_in[19];
    const float* Wff2  = (const float*)d_in[20];
    const float* bff2  = (const float*)d_in[21];
    const float* Wlog  = (const float*)d_in[22];
    const float* blog  = (const float*)d_in[23];

    float* logits = (float*)d_out;
    float* aux    = logits + (long)BB * NNc * VV;

    // ---- workspace layout (256B aligned)
    char* wsb = (char*)d_ws;
    size_t off = 0;
    auto alloc = [&](size_t bytes) -> void* {
        void* p = wsb + off;
        off = (off + bytes + 255) & ~(size_t)255;
        return p;
    };
    __bf16* wq_bf   = (__bf16*)alloc((size_t)LL * DD * DD * 2);
    __bf16* wkv_bf  = (__bf16*)alloc((size_t)LL * DD * 2 * DD * 2);
    __bf16* wo_bf   = (__bf16*)alloc((size_t)LL * DD * DD * 2);
    __bf16* wpos_bf = (__bf16*)alloc((size_t)LL * DD * DHH * 2);
    __bf16* wff1_bf = (__bf16*)alloc((size_t)LL * DD * FFD * 2);
    __bf16* wff2_bf = (__bf16*)alloc((size_t)LL * FFD * DD * 2);
    __bf16* wlog_bf = (__bf16*)alloc((size_t)DD * VV * 2);
    float*  h       = (float*) alloc((size_t)BB * NNc * DD * 4);
    __bf16* xn_bf   = (__bf16*)alloc((size_t)BB * NNc * DD * 2);
    __bf16* ctx_bf  = (__bf16*)alloc((size_t)BB * CTX * DD * 2);
    __bf16* q_bf    = (__bf16*)alloc((size_t)BB * NNc * DD * 2);
    __bf16* kv_bf   = (__bf16*)alloc((size_t)BB * CTX * 2 * DD * 2);
    __bf16* pe_bf   = (__bf16*)alloc((size_t)NNc * DD * 2);
    __bf16* pos_bf  = (__bf16*)alloc((size_t)NNc * DHH * 2);
    __bf16* posT_bf = (__bf16*)alloc((size_t)DHH * NNc * 2);
    float*  qpos    = (float*) alloc((size_t)BB * HH * NNc * NNc * 4);
    float*  emask   = (float*) alloc((size_t)BB * CTX * 4);
    __bf16* out_bf  = (__bf16*)alloc((size_t)BB * NNc * DD * 2);
    __bf16* ff_bf   = (__bf16*)alloc((size_t)BB * NNc * FFD * 2);
    __bf16* h_bf    = (__bf16*)alloc((size_t)BB * NNc * DD * 2);
    (void)ws_size; (void)in_sizes; (void)n_in; (void)out_size;

    auto cast = [&](const float* s, __bf16* d, long n) {
        k_cast<<<dim3((unsigned)((n / 4 + 255) / 256)), dim3(256), 0, stream>>>(s, d, n);
    };
    auto gemm = [&](const __bf16* A, long lda, const __bf16* W, long ldw,
                    const float* bias, const float* resid, long ldr,
                    float* Cf, __bf16* Cb, long ldc,
                    long Mr, long Nc, int K, int gelu,
                    int nbatch = 1, int batchH = 0, long sAb = 0, long sAh = 0, long sC = 0) {
        dim3 grid((unsigned)(Nc / 64), (unsigned)(Mr / 128), (unsigned)nbatch);
        k_gemm<<<grid, dim3(256), 0, stream>>>(A, lda, W, ldw, bias, resid, ldr,
                                               Cf, Cb, ldc, K, gelu, batchH, sAb, sAh, sC);
    };

    // ---- one-time per call: weight casts, aux zero, embedding, sinusoid PE
    cast(Wq,   wq_bf,   (long)LL * DD * DD);
    cast(Wkv,  wkv_bf,  (long)LL * DD * 2 * DD);
    cast(Wo,   wo_bf,   (long)LL * DD * DD);
    cast(Wpos, wpos_bf, (long)LL * DD * DHH);
    cast(Wff1, wff1_bf, (long)LL * DD * FFD);
    cast(Wff2, wff2_bf, (long)LL * FFD * DD);
    cast(Wlog, wlog_bf, (long)DD * VV);
    k_zero1<<<1, 32, 0, stream>>>(aux);
    k_embed<<<dim3(BB * NNc), dim3(256), 0, stream>>>(x, emb, h);
    k_sinusoid<<<dim3(NNc), dim3(256), 0, stream>>>(pe_bf);

    for (int l = 0; l < LL; ++l) {
        const float* mem_l = mems + (size_t)l * BB * MMc * DD;
        // expire-span masks + aux
        k_exps<<<dim3(BB * CTX), dim3(256), 0, stream>>>(
            mem_l, times + (size_t)l * BB * MMc, Wexp + (size_t)l * DD, bexp, l, emask, aux);
        // xn = LN1(h)
        k_layernorm<<<dim3(BB * NNc), dim3(256), 0, stream>>>(
            h, ln1g + (size_t)l * DD, ln1b + (size_t)l * DD, xn_bf);
        // ctx = [mem ; xn]
        k_ctx<<<dim3(BB * CTX), dim3(256), 0, stream>>>(mem_l, xn_bf, ctx_bf);
        // q = xn @ Wq + bq   (bf16 out)
        gemm(xn_bf, DD, wq_bf + (size_t)l * DD * DD, DD, bq + (size_t)l * DD,
             nullptr, 0, nullptr, q_bf, DD, (long)BB * NNc, DD, DD, 0);
        // kv = ctx @ Wkv + bkv  (bf16 out, k|v packed on last dim)
        gemm(ctx_bf, DD, wkv_bf + (size_t)l * DD * 2 * DD, 2 * DD, bkv + (size_t)l * 2 * DD,
             nullptr, 0, nullptr, kv_bf, 2 * DD, (long)BB * CTX, 2 * DD, DD, 0);
        // pos = pe @ Wpos + bpos (bf16), then transpose for use as GEMM B
        gemm(pe_bf, DD, wpos_bf + (size_t)l * DD * DHH, DHH, bpos + (size_t)l * DHH,
             nullptr, 0, nullptr, pos_bf, DHH, (long)NNc, DHH, DD, 0);
        k_posT<<<dim3(NNc / 256, DHH), dim3(256), 0, stream>>>(pos_bf, posT_bf);
        // qpos[b,h] = q[b,:,h*64:+64] @ pos^T  (batched over B*H)
        gemm(q_bf, DD, posT_bf, NNc, nullptr, nullptr, 0, qpos, nullptr, NNc,
             (long)NNc, (long)NNc, DHH, 0,
             BB * HH, HH, (long)NNc * DD, 64, (long)NNc * NNc);
        // flash attention -> out_bf
        k_attn<<<dim3(NNc / 64, BB * HH), dim3(128), 0, stream>>>(
            q_bf, kv_bf, qpos, emask, out_bf);
        // h = out @ Wo + bo + h
        gemm(out_bf, DD, wo_bf + (size_t)l * DD * DD, DD, bo + (size_t)l * DD,
             h, DD, h, nullptr, DD, (long)BB * NNc, DD, DD, 0);
        // hn = LN2(h)
        k_layernorm<<<dim3(BB * NNc), dim3(256), 0, stream>>>(
            h, ln2g + (size_t)l * DD, ln2b + (size_t)l * DD, xn_bf);
        // ff = gelu(hn @ Wff1 + bff1)
        gemm(xn_bf, DD, wff1_bf + (size_t)l * DD * FFD, FFD, bff1 + (size_t)l * FFD,
             nullptr, 0, nullptr, ff_bf, FFD, (long)BB * NNc, FFD, DD, 1);
        // h = ff @ Wff2 + bff2 + h
        gemm(ff_bf, FFD, wff2_bf + (size_t)l * FFD * DD, DD, bff2 + (size_t)l * DD,
             h, DD, h, nullptr, DD, (long)BB * NNc, DD, FFD, 0);
    }

    // logits = h @ Wlog + blog  (fp32 into d_out)
    cast(h, h_bf, (long)BB * NNc * DD);
    gemm(h_bf, DD, wlog_bf, VV, blog, nullptr, 0, logits, nullptr, VV,
         (long)BB * NNc, VV, DD, 0);
}
